// Decoder_45148696215569
// MI455X (gfx1250) — matrix operand order repaired and checked by reading, not compiled, not run on hardware
//
#include <hip/hip_runtime.h>
#include <hip/hip_bf16.h>

typedef __attribute__((ext_vector_type(16))) __bf16 v16bf;
typedef __attribute__((ext_vector_type(4)))  __bf16 bf4;
typedef __attribute__((ext_vector_type(8)))  float  v8f;
typedef __attribute__((ext_vector_type(4), aligned(4))) float f4a;

#define B_    8
#define C_    96
#define H_    64
#define W_    128
#define PREVC 661
#define FEATC 629

// fragment-ready weight LDS layout: per cout, 9 K-steps x 2 half-wave groups x 16 bf16 (+16 pad)
#define WSTR  304   // 288 + 16 pad elements per cout row (608 B, odd multiple of 32 B)

__device__ __forceinline__ float leakyf(float v) { return v >= 0.f ? v : 0.1f * v; }

// -------------------- conv_transpose2d(k=4, s=2, p=1), Cout=2 --------------------
__global__ void upconv_kernel(const float* __restrict__ in, const float* __restrict__ w,
                              const float* __restrict__ bias, float* __restrict__ feat,
                              int Cin, int chOff)
{
    int idx = blockIdx.x * blockDim.x + threadIdx.x;
    if (idx >= B_ * 2 * H_ * W_) return;
    int x  = idx % W_;
    int y  = (idx / W_) % H_;
    int co = (idx / (W_ * H_)) % 2;
    int b  = idx / (W_ * H_ * 2);
    const int H2 = H_ / 2, W2 = W_ / 2;

    int kys[2], iys[2], nky = 0;
    for (int ky = 0; ky < 4; ++ky) {
        int t = y + 1 - ky;
        if (t >= 0 && (t & 1) == 0 && (t >> 1) < H2) { kys[nky] = ky; iys[nky] = t >> 1; ++nky; }
    }
    int kxs[2], ixs[2], nkx = 0;
    for (int kx = 0; kx < 4; ++kx) {
        int t = x + 1 - kx;
        if (t >= 0 && (t & 1) == 0 && (t >> 1) < W2) { kxs[nkx] = kx; ixs[nkx] = t >> 1; ++nkx; }
    }
    float acc = bias[co];
    for (int ci = 0; ci < Cin; ++ci) {
        const float* ip = in + (size_t)(b * Cin + ci) * H2 * W2;
        const float* wp = w + (size_t)(ci * 2 + co) * 16;
        for (int a = 0; a < nky; ++a)
            for (int c2 = 0; c2 < nkx; ++c2)
                acc += ip[iys[a] * W2 + ixs[c2]] * wp[kys[a] * 4 + kxs[c2]];
    }
    feat[(((size_t)b * FEATC + chOff + co) * H_ + y) * W_ + x] = acc;
}

// -------------------- copy tenOne into feat channels [529, 625) --------------------
__global__ void copy_one_kernel(const float* __restrict__ one, float* __restrict__ feat)
{
    size_t idx = (size_t)blockIdx.x * blockDim.x + threadIdx.x;
    const size_t per = (size_t)C_ * H_ * W_;
    if (idx >= (size_t)B_ * per) return;
    size_t b = idx / per, r = idx % per;
    feat[((size_t)b * FEATC + 529) * (size_t)(H_ * W_) + r] = one[idx];
}

// -------------------- bilinear backwarp of tenTwo by 1.25 * upflow --------------------
__global__ void warp_kernel(const float* __restrict__ two, const float* __restrict__ feat,
                            float* __restrict__ warped)
{
    int idx = blockIdx.x * blockDim.x + threadIdx.x;
    if (idx >= B_ * C_ * H_ * W_) return;
    int x = idx % W_, y = (idx / W_) % H_;
    int c = (idx / (W_ * H_)) % C_, b = idx / (W_ * H_ * C_);
    size_t fb = (size_t)b * FEATC * (size_t)(H_ * W_);
    float fx = feat[fb + (size_t)625 * H_ * W_ + y * W_ + x] * 1.25f;
    float fy = feat[fb + (size_t)626 * H_ * W_ + y * W_ + x] * 1.25f;
    float px = x + fx, py = y + fy;
    float x0f = floorf(px), y0f = floorf(py);
    int x0 = (int)x0f, y0 = (int)y0f;
    float wx = px - x0f, wy = py - y0f;
    const float* fp = two + (size_t)(b * C_ + c) * H_ * W_;
    float s = 0.f;
#pragma unroll
    for (int dy = 0; dy < 2; ++dy)
#pragma unroll
        for (int dx = 0; dx < 2; ++dx) {
            int yi = y0 + dy, xi = x0 + dx;
            float wgt = (dy ? wy : 1.f - wy) * (dx ? wx : 1.f - wx);
            if (yi >= 0 && yi < H_ && xi >= 0 && xi < W_) s += wgt * fp[yi * W_ + xi];
        }
    warped[idx] = s;
}

// -------------------- 81-tap cost volume (mean over C), leaky, into feat [448,529) ----
__global__ void corr_kernel(const float* __restrict__ one, const float* __restrict__ warped,
                            float* __restrict__ feat)
{
    int idx = blockIdx.x * blockDim.x + threadIdx.x;
    if (idx >= B_ * 81 * H_ * W_) return;
    int x = idx % W_, y = (idx / W_) % H_;
    int t = (idx / (W_ * H_)) % 81, b = idx / (W_ * H_ * 81);
    int dy = t / 9 - 4, dx = t % 9 - 4;
    int yy = y + dy, xx = x + dx;
    float s = 0.f;
    if (yy >= 0 && yy < H_ && xx >= 0 && xx < W_) {
        const float* p1 = one + ((size_t)b * C_ * H_ + y) * W_ + x;
        const float* p2 = warped + ((size_t)b * C_ * H_ + yy) * W_ + xx;
        for (int c = 0; c < C_; ++c)
            s += p1[(size_t)c * H_ * W_] * p2[(size_t)c * H_ * W_];
        s *= (1.f / 96.f);
    }
    feat[(((size_t)b * FEATC + 448 + t) * H_ + y) * W_ + x] = leakyf(s);
}

// -------------------- implicit-GEMM 3x3 conv via WMMA bf16 --------------------
// Block: 256 threads = 8 wave32. Tile: 64 Cout x 64 pixels (one row segment).
// Wave (m_sub, n_pair): m_sub in 0..3 picks 16 couts; n_pair in 0..1 picks 32 pixels,
// computed as two 16x16 subtiles sharing one A fragment (2 accumulators / wave).
// K-chunk = 32 input channels -> 288 k-values = 9 WMMA K-steps of 32.
__global__ __launch_bounds__(256)
void conv3x3_wmma_kernel(const float* __restrict__ feat, int inOff, int Cin,
                         const float* __restrict__ w, const float* __restrict__ bias,
                         float* __restrict__ out, int outOff, int outCtot, int Cout)
{
    __shared__ alignas(32) __bf16 lds_w[64 * WSTR];   // fragment-ready weights (38.9 KB)
    __shared__ alignas(32) __bf16 lds_in[32 * 198];   // 32 ch x 3 rows x 66 px (12.7 KB)

    const int tid  = threadIdx.x;
    const int lane = tid & 31;
    const int wave = tid >> 5;
    const int m_sub  = wave & 3;
    const int n_pair = wave >> 2;

    const int XT = W_ / 64;
    int bi = blockIdx.x;
    int xt = bi % XT; bi /= XT;
    int y  = bi % H_;
    int b  = bi / H_;
    int x0 = xt * 64;
    int coutBase = blockIdx.y * 64;

    const int m  = lane & 15;
    const int hi = lane >> 4;

    v8f acc0 = {}, acc1 = {};

    for (int c0 = 0; c0 < Cin; c0 += 32) {
        int kvalid = (Cin - c0 < 32 ? Cin - c0 : 32) * 9;
        __syncthreads();

        // ---- stage weights (64 couts x 288 k) into fragment-ready order ----
        // linear kk -> step s = kk/32; within step k32: hi = k32[3],
        // e = (k32[4]<<3) | k32[2:0]; slot = cl*WSTR + (s*2+hi)*16 + e.
        // 4 consecutive kk (kk%4==0) map to 4 consecutive e -> bf16x4 ds_store_b64.
        for (int q = tid; q < 64 * 72; q += 256) {   // 72 = 288/4 quads per cout
            int cl = q / 72;
            int kk = (q - cl * 72) * 4;
            int co = coutBase + cl;
            f4a v = {0.f, 0.f, 0.f, 0.f};
            if (co < Cout) {
                const float* wp = w + (size_t)co * (size_t)Cin * 9 + (size_t)c0 * 9 + kk;
                if (kk + 3 < kvalid) {
                    v = *(const f4a*)wp;
                } else {
#pragma unroll
                    for (int j = 0; j < 4; ++j)
                        if (kk + j < kvalid) v[j] = wp[j];
                }
            }
            int s   = kk >> 5;
            int k32 = kk & 31;
            int e0  = ((k32 >> 4) << 3) | (k32 & 7);
            bf4 pk = { (__bf16)v[0], (__bf16)v[1], (__bf16)v[2], (__bf16)v[3] };
            *(bf4*)&lds_w[cl * WSTR + ((s * 2 + ((k32 >> 3) & 1)) * 16) + e0] = pk;
        }
        // ---- stage activation halo tile (32 ch x 3 rows x 66 px), zero padded ----
        for (int idx = tid; idx < 32 * 198; idx += 256) {
            int cc  = idx / 198;
            int rem = idx - cc * 198;
            int r  = rem / 66, xx = rem - r * 66;
            int c  = c0 + cc, ys = y + r - 1, xs = x0 + xx - 1;
            float v = 0.f;
            if (c < Cin && (unsigned)ys < (unsigned)H_ && (unsigned)xs < (unsigned)W_)
                v = feat[(((size_t)b * FEATC + inOff + c) * H_ + ys) * W_ + xs];
            lds_in[idx] = (__bf16)v;
        }
        if (c0 + 32 < Cin)
            __builtin_prefetch(&feat[(((size_t)b * FEATC + inOff + c0 + 32) * H_ + y) * W_ + x0], 0, 1);
        __syncthreads();

#pragma unroll
        for (int s = 0; s < 9; ++s) {
            // A fragment: one aligned 32-byte LDS load (2 x ds_load_b128)
            v16bf af = *(const v16bf*)&lds_w[(m_sub * 16 + m) * WSTR + (s * 2 + hi) * 16];

            // B fragments: lane covers pixel m; element e -> K-row s*32 + table-order k(hi, e)
            v16bf b0, b1;
#pragma unroll
            for (int e = 0; e < 16; ++e) {
                int kB = s * 32 + (e < 8 ? (hi << 3) + e : 16 + (hi << 3) + (e - 8));
                int cc = kB / 9, tap = kB - cc * 9;
                int ky = tap / 3, kx = tap - ky * 3;
                const __bf16* src = &lds_in[cc * 198 + ky * 66 + n_pair * 32 + kx + m];
                b0[e] = src[0];
                b1[e] = src[16];
            }

            acc0 = __builtin_amdgcn_wmma_f32_16x16x32_bf16(false, af, false, b0,
                                                           (short)0, acc0, false, false);
            acc1 = __builtin_amdgcn_wmma_f32_16x16x32_bf16(false, af, false, b1,
                                                           (short)0, acc1, false, false);
        }
    }

    // C/D layout: lane -> N (pixel), VGPR r -> M = r + 8*hi (cout)
    int xo0 = x0 + n_pair * 32 + (lane & 15);
#pragma unroll
    for (int r = 0; r < 8; ++r) {
        int co = coutBase + m_sub * 16 + r + (hi << 3);
        if (co < Cout) {
            float bv = bias[co];
            size_t base = (((size_t)b * outCtot + outOff + co) * H_ + y) * W_;
            float v0 = acc0[r] + bv;
            float v1 = acc1[r] + bv;
            out[base + xo0]      = leakyf(v0);
            out[base + xo0 + 16] = leakyf(v1);
        }
    }
}

extern "C" void kernel_launch(void* const* d_in, const int* in_sizes, int n_in,
                              void* d_out, int out_size, void* d_ws, size_t ws_size,
                              hipStream_t stream)
{
    (void)in_sizes; (void)n_in; (void)out_size; (void)ws_size;
    const float* tenOne   = (const float*)d_in[0];
    const float* tenTwo   = (const float*)d_in[1];
    const float* prevFlow = (const float*)d_in[2];
    const float* prevFeat = (const float*)d_in[3];
    const float* wUpflow  = (const float*)d_in[4];
    const float* bUpflow  = (const float*)d_in[5];
    const float* wUpfeat  = (const float*)d_in[6];
    const float* bUpfeat  = (const float*)d_in[7];
    const float* wconv[6]; const float* bconv[6];
    for (int i = 0; i < 6; ++i) {
        wconv[i] = (const float*)d_in[8 + 2 * i];
        bconv[i] = (const float*)d_in[9 + 2 * i];
    }

    float* flowOut = (float*)d_out;                       // (B,2,H,W)
    float* feat    = flowOut + (size_t)B_ * 2 * H_ * W_;  // (B,629,H,W) final feat
    float* warped  = (float*)d_ws;                        // (B,96,H,W) scratch

    const int TPB = 256;
    int n_up = B_ * 2 * H_ * W_;
    upconv_kernel<<<(n_up + TPB - 1) / TPB, TPB, 0, stream>>>(prevFlow, wUpflow, bUpflow, feat, 2, 625);
    upconv_kernel<<<(n_up + TPB - 1) / TPB, TPB, 0, stream>>>(prevFeat, wUpfeat, bUpfeat, feat, PREVC, 627);

    int n_cp = B_ * C_ * H_ * W_;
    copy_one_kernel<<<(n_cp + TPB - 1) / TPB, TPB, 0, stream>>>(tenOne, feat);
    warp_kernel<<<(n_cp + TPB - 1) / TPB, TPB, 0, stream>>>(tenTwo, feat, warped);

    int n_corr = B_ * 81 * H_ * W_;
    corr_kernel<<<(n_corr + TPB - 1) / TPB, TPB, 0, stream>>>(tenOne, warped, feat);

    struct L { int inOff, Cin, Cout, outOff; };
    const L layers[5] = {
        {448, 181, 128, 320},
        {320, 309, 128, 192},
        {192, 437,  96,  96},
        { 96, 533,  64,  32},
        { 32, 597,  32,   0},
    };
    dim3 blk(256);
    for (int i = 0; i < 5; ++i) {
        dim3 grid(B_ * H_ * (W_ / 64), (layers[i].Cout + 63) / 64);
        conv3x3_wmma_kernel<<<grid, blk, 0, stream>>>(feat, layers[i].inOff, layers[i].Cin,
                                                      wconv[i], bconv[i],
                                                      feat, layers[i].outOff, FEATC, layers[i].Cout);
    }
    // conv6: 629 -> 2 (flow), leaky per reference (netSix is a ConvBlock)
    dim3 grid6(B_ * H_ * (W_ / 64), 1);
    conv3x3_wmma_kernel<<<grid6, blk, 0, stream>>>(feat, 0, 629, wconv[5], bconv[5],
                                                   flowOut, 0, 2, 2);
}
